// GPT_86191403696522
// MI455X (gfx1250) — matrix operand order repaired and checked by reading, not compiled, not run on hardware
//
#include <hip/hip_runtime.h>
#include <hip/hip_bf16.h>

// ---------------- Model config (matches reference) ----------------
#define T_SEQ  2048
#define HID    1024
#define VOCAB  32000
#define NHEAD  16
#define NKVH   4
#define HDIM   64
#define FFD    4096
#define NLAYER 4
#define KVD    256     // NKVH * HDIM

typedef __bf16 bf16;
typedef __attribute__((ext_vector_type(16))) __bf16 v16bf;
typedef __attribute__((ext_vector_type(8)))  float  v8f;

// ---------------- small device helpers ----------------
__device__ inline v8f v8f_zero() {
    v8f z;
#pragma unroll
    for (int j = 0; j < 8; ++j) z[j] = 0.0f;
    return z;
}

__device__ inline float red_max16(float x) {
#pragma unroll
    for (int m = 1; m < 16; m <<= 1) x = fmaxf(x, __shfl_xor(x, m));
    return x;
}
__device__ inline float red_sum16(float x) {
#pragma unroll
    for (int m = 1; m < 16; m <<= 1) x += __shfl_xor(x, m);
    return x;
}
__device__ inline float wave_sum32(float x) {
#pragma unroll
    for (int m = 1; m < 32; m <<= 1) x += __shfl_xor(x, m);
    return x;
}

// async global -> LDS copy of 16B, per-lane addresses (ASYNCcnt tracked)
__device__ inline void async_g2l_b128(unsigned lds_addr, const void* gaddr) {
    unsigned long long ga = (unsigned long long)(uintptr_t)gaddr;
    asm volatile("global_load_async_to_lds_b128 %0, %1, off"
                 :: "v"(lds_addr), "v"(ga) : "memory");
}
__device__ inline void wait_asynccnt0() {
    asm volatile("s_wait_asynccnt 0x0" ::: "memory");
}

// ---------------- f32 -> bf16 conversion ----------------
__global__ void convert_kernel(const float* __restrict__ in, bf16* __restrict__ out, long n) {
    long i = (long)blockIdx.x * blockDim.x + threadIdx.x;
    long stride = (long)gridDim.x * blockDim.x;
    for (; i < n; i += stride) out[i] = (bf16)in[i];
}

// ---------------- token + position embedding ----------------
__global__ void embed_kernel(const int* __restrict__ ids, const float* __restrict__ tok,
                             const float* __restrict__ pos, float* __restrict__ x) {
    int t = blockIdx.x;
    int id = ids[t];
    for (int i = threadIdx.x; i < HID; i += blockDim.x)
        x[(size_t)t * HID + i] = tok[(size_t)id * HID + i] + pos[(size_t)t * HID + i];
}

// ---------------- LayerNorm: f32 in -> bf16 out ----------------
__global__ __launch_bounds__(256)
void layernorm_kernel(const float* __restrict__ x, const float* __restrict__ g,
                      const float* __restrict__ b, bf16* __restrict__ out) {
    __shared__ float sh[8];
    const int row  = blockIdx.x;
    const int lane = threadIdx.x & 31;
    const int wave = threadIdx.x >> 5;
    const float* xr = x + (size_t)row * HID;

    float s = 0.0f;
    for (int i = threadIdx.x; i < HID; i += 256) s += xr[i];
    s = wave_sum32(s);
    if (lane == 0) sh[wave] = s;
    __syncthreads();
    float tot = 0.0f;
#pragma unroll
    for (int w = 0; w < 8; ++w) tot += sh[w];
    const float mean = tot * (1.0f / HID);
    __syncthreads();

    float vs = 0.0f;
    for (int i = threadIdx.x; i < HID; i += 256) {
        float d = xr[i] - mean;
        vs += d * d;
    }
    vs = wave_sum32(vs);
    if (lane == 0) sh[wave] = vs;
    __syncthreads();
    float vtot = 0.0f;
#pragma unroll
    for (int w = 0; w < 8; ++w) vtot += sh[w];
    const float rstd = rsqrtf(vtot * (1.0f / HID) + 1e-5f);

    for (int i = threadIdx.x; i < HID; i += 256)
        out[(size_t)row * HID + i] = (bf16)((xr[i] - mean) * rstd * g[i] + b[i]);
}

// ---------------- Tiled bf16 WMMA GEMM, double-buffered, async A-stage -------
// C[M,N] = act(A[M,K] @ B[K,N] + bias) [+ residual]; write f32 and/or bf16.
// block = 256 threads (8 waves); tile 128(M) x 128(N); K-step 32.
// Waves: 2 (M) x 4 (N); wave tile 64x32 -> 4x2 WMMA accumulators.
#define TM 128
#define TN 128
#define TK 32
#define TLD 48    // LDS row stride in bf16 (96B: 16B-aligned, spreads banks)

template<bool HAS_BIAS, bool GELU, bool HAS_RES, bool OUTF, bool OUTB>
__global__ __launch_bounds__(256)
void gemm_bf16_kernel(const bf16* __restrict__ A, const bf16* __restrict__ B,
                      const float* __restrict__ bias, const float* __restrict__ residual,
                      float* __restrict__ outF, bf16* __restrict__ outB,
                      int M, int N, int K) {
    __shared__ bf16 As[2][TM][TLD];     // A tile   [m][k]
    __shared__ bf16 BsT[2][TN][TLD];    // B tile transposed [n][k]

    const int tid  = threadIdx.x;
    const int lane = tid & 31;
    const int wave = tid >> 5;       // 0..7
    const int wm   = wave & 1;       // M sub-tile: wm*64
    const int wn   = wave >> 1;      // N sub-tile: wn*32
    const int blockM = blockIdx.y * TM;
    const int blockN = blockIdx.x * TN;
    const int m    = lane & 15;
    const int half = (lane >> 4) & 1;

    // staging geometry
    const int arow  = tid >> 1;            // A: 0..127
    const int ahalf = (tid & 1) * 16;      // A: k-chunk 0 or 16
    const int brow  = tid >> 3;            // B: k index 0..31
    const int bcol  = (tid & 7) * 16;      // B: col start (16 cols/thread)

    v8f acc[4][2];
#pragma unroll
    for (int i = 0; i < 4; ++i)
#pragma unroll
        for (int j = 0; j < 2; ++j) acc[i][j] = v8f_zero();

    // ---- prologue: stage tile 0 into buffer 0 ----
    {
        const bf16* asrc = A + (size_t)(blockM + arow) * K + ahalf;
        unsigned la = (unsigned)(uintptr_t)&As[0][arow][ahalf];
        async_g2l_b128(la, asrc);
        async_g2l_b128(la + 16, asrc + 8);

        const bf16* bsrc = B + (size_t)brow * N + blockN + bcol;
        union { uint4 u[2]; bf16 h[16]; } bu;
        bu.u[0] = ((const uint4*)bsrc)[0];
        bu.u[1] = ((const uint4*)bsrc)[1];
#pragma unroll
        for (int j = 0; j < 16; ++j) BsT[0][bcol + j][brow] = bu.h[j];
    }
    wait_asynccnt0();
    __syncthreads();

    int buf = 0;
    for (int k0 = 0; k0 < K; k0 += TK, buf ^= 1) {
        const bool has_next = (k0 + TK) < K;

        // ---- issue next tile's global traffic early ----
        union { uint4 u[2]; bf16 h[16]; } bu;
        if (has_next) {
            const bf16* asrc = A + (size_t)(blockM + arow) * K + (k0 + TK) + ahalf;
            unsigned la = (unsigned)(uintptr_t)&As[buf ^ 1][arow][ahalf];
            async_g2l_b128(la, asrc);
            async_g2l_b128(la + 16, asrc + 8);

            const bf16* bsrc = B + (size_t)(k0 + TK + brow) * N + blockN + bcol;
            bu.u[0] = ((const uint4*)bsrc)[0];
            bu.u[1] = ((const uint4*)bsrc)[1];
            if (k0 + 2 * TK < K) __builtin_prefetch(bsrc + (size_t)TK * N, 0, 1);
        }

        // ---- fragments from current buffer (A and B both in contiguous k = 16*half + s order) ----
        v16bf afrag[4], bfrag[2];
        const int koffA = half ? 16 : 0;
#pragma unroll
        for (int mi = 0; mi < 4; ++mi) {
            const bf16* rowp = &As[buf][wm * 64 + mi * 16 + m][0];
#pragma unroll
            for (int j = 0; j < 8; ++j) {
                afrag[mi][j]     = rowp[koffA + j];
                afrag[mi][8 + j] = rowp[koffA + 8 + j];
            }
        }
        const int kb = half ? 16 : 0;
#pragma unroll
        for (int ni = 0; ni < 2; ++ni) {
            const bf16* colp = &BsT[buf][wn * 32 + ni * 16 + m][0];
#pragma unroll
            for (int j = 0; j < 16; ++j) bfrag[ni][j] = colp[kb + j];
        }

#pragma unroll
        for (int mi = 0; mi < 4; ++mi)
#pragma unroll
            for (int ni = 0; ni < 2; ++ni)
                acc[mi][ni] = __builtin_amdgcn_wmma_f32_16x16x32_bf16(
                    false, afrag[mi], false, bfrag[ni], (short)0, acc[mi][ni], false, false);

        // ---- land next B tile (transposed) into the other buffer ----
        if (has_next) {
#pragma unroll
            for (int j = 0; j < 16; ++j) BsT[buf ^ 1][bcol + j][brow] = bu.h[j];
        }
        wait_asynccnt0();
        __syncthreads();
    }

    // ---- epilogue: C-layout row = r + 8*half, col = lane%16 ----
#pragma unroll
    for (int mi = 0; mi < 4; ++mi) {
#pragma unroll
        for (int ni = 0; ni < 2; ++ni) {
            const int gcol = blockN + wn * 32 + ni * 16 + m;
            const float bv = HAS_BIAS ? bias[gcol] : 0.0f;
#pragma unroll
            for (int r = 0; r < 8; ++r) {
                const int grow = blockM + wm * 64 + mi * 16 + r + 8 * half;
                float v = acc[mi][ni][r] + bv;
                if (GELU) v = 0.5f * v * (1.0f + erff(v * 0.70710678118f));
                const size_t idx = (size_t)grow * N + gcol;
                if (HAS_RES) v += residual[idx];
                if (OUTF) outF[idx] = v;
                if (OUTB) outB[idx] = (bf16)v;
            }
        }
    }
}

// ---------------- Flash-attention (GQA, full attention, additive key mask) ----
// grid (T/64, NHEAD), block 128 (4 waves). Each wave: 16 query rows.
__global__ __launch_bounds__(128)
void attn_kernel(const bf16* __restrict__ q, const bf16* __restrict__ k,
                 const bf16* __restrict__ v, const float* __restrict__ mask,
                 bf16* __restrict__ out) {
    __shared__ bf16 Plds[4][16][32];
    const int lane = threadIdx.x & 31;
    const int wave = threadIdx.x >> 5;    // 0..3
    const int qh   = blockIdx.y;          // query head
    const int kvh  = qh >> 2;             // GQA group of 4
    const int qrow0 = blockIdx.x * 64 + wave * 16;
    const int m    = lane & 15;
    const int half = (lane >> 4) & 1;

    // Q fragments (A-layout) for d 0..31, 32..63
    const bf16* qrow = q + (size_t)(qrow0 + m) * HID + qh * HDIM;
    const int koff = half ? 16 : 0;
    v16bf qf[2];
#pragma unroll
    for (int di = 0; di < 2; ++di)
#pragma unroll
        for (int j = 0; j < 8; ++j) {
            qf[di][j]     = qrow[di * 32 + koff + j];
            qf[di][8 + j] = qrow[di * 32 + koff + 8 + j];
        }

    float mrow[8], lrow[8], alpha[8];
    v8f O[4];
#pragma unroll
    for (int r = 0; r < 8; ++r) { mrow[r] = -1e30f; lrow[r] = 0.0f; }
#pragma unroll
    for (int n = 0; n < 4; ++n) O[n] = v8f_zero();

    const int kbB = half ? 16 : 0;

    for (int s0 = 0; s0 < T_SEQ; s0 += 32) {
        // ---- S = Q K^T for 32 keys (two 16-key WMMA column tiles) ----
        v8f S[2];
#pragma unroll
        for (int sh = 0; sh < 2; ++sh) {
            const bf16* kcol = k + (size_t)(s0 + sh * 16 + m) * KVD + kvh * HDIM;
            v16bf kf0, kf1;
#pragma unroll
            for (int j = 0; j < 16; ++j) {
                kf0[j] = kcol[kbB + j];        // d 0..31 tile
                kf1[j] = kcol[32 + kbB + j];   // d 32..63 tile
            }
            v8f s_acc = v8f_zero();
            s_acc = __builtin_amdgcn_wmma_f32_16x16x32_bf16(false, qf[0], false, kf0,
                                                            (short)0, s_acc, false, false);
            s_acc = __builtin_amdgcn_wmma_f32_16x16x32_bf16(false, qf[1], false, kf1,
                                                            (short)0, s_acc, false, false);
            S[sh] = s_acc;
        }

        const float mk0 = mask[s0 + m];
        const float mk1 = mask[s0 + 16 + m];

        // ---- online softmax update per row ----
#pragma unroll
        for (int r = 0; r < 8; ++r) {
            float x0 = S[0][r] * 0.125f + mk0;   // 1/sqrt(64)
            float x1 = S[1][r] * 0.125f + mk1;
            float tmax = red_max16(fmaxf(x0, x1));
            float mnew = fmaxf(mrow[r], tmax);
            float p0 = __expf(x0 - mnew);
            float p1 = __expf(x1 - mnew);
            float rs = red_sum16(p0 + p1);
            float a = __expf(mrow[r] - mnew);
            lrow[r] = lrow[r] * a + rs;
            mrow[r] = mnew;
            alpha[r] = a;
            const int row = r + 8 * half;
            Plds[wave][row][m]      = (bf16)p0;
            Plds[wave][row][16 + m] = (bf16)p1;
        }
#pragma unroll
        for (int n = 0; n < 4; ++n)
#pragma unroll
            for (int r = 0; r < 8; ++r) O[n][r] *= alpha[r];

        // wave-local LDS RAW: C-layout stores -> A-layout loads
        asm volatile("s_wait_dscnt 0x0" ::: "memory");

        v16bf pf;
        {
            const bf16* prow = &Plds[wave][m][0];
#pragma unroll
            for (int j = 0; j < 8; ++j) {
                pf[j]     = prow[koff + j];
                pf[8 + j] = prow[koff + 8 + j];
            }
        }

        // ---- O += P @ V (4 d-tiles of 16) ----
#pragma unroll
        for (int n = 0; n < 4; ++n) {
            const bf16* vp = v + (size_t)(s0 + kbB) * KVD + kvh * HDIM + n * 16 + m;
            v16bf vf;
#pragma unroll
            for (int j = 0; j < 16; ++j) vf[j] = vp[(size_t)j * KVD];
            O[n] = __builtin_amdgcn_wmma_f32_16x16x32_bf16(false, pf, false, vf,
                                                           (short)0, O[n], false, false);
        }
    }

    // ---- normalize + store bf16 ----
#pragma unroll
    for (int n = 0; n < 4; ++n)
#pragma unroll
        for (int r = 0; r < 8; ++r) {
            const int row = r + 8 * half;
            const int t = qrow0 + row;
            out[(size_t)t * HID + qh * HDIM + n * 16 + m] = (bf16)(O[n][r] / lrow[r]);
        }
}

// ---------------- host-side orchestration ----------------
extern "C" void kernel_launch(void* const* d_in, const int* in_sizes, int n_in,
                              void* d_out, int out_size, void* d_ws, size_t ws_size,
                              hipStream_t stream) {
    (void)in_sizes; (void)n_in; (void)out_size; (void)ws_size;

    const int*   input_ids = (const int*)  d_in[0];
    const float* attn_mask = (const float*)d_in[1];
    const float* tok_emb   = (const float*)d_in[2];
    const float* pos_emb   = (const float*)d_in[3];
    const float* qW  = (const float*)d_in[4];   const float* qb  = (const float*)d_in[5];
    const float* kW  = (const float*)d_in[6];   const float* kb  = (const float*)d_in[7];
    const float* vW  = (const float*)d_in[8];   const float* vb  = (const float*)d_in[9];
    const float* oW  = (const float*)d_in[10];  const float* ob  = (const float*)d_in[11];
    const float* m1W = (const float*)d_in[12];  const float* m1b = (const float*)d_in[13];
    const float* m2W = (const float*)d_in[14];  const float* m2b = (const float*)d_in[15];
    const float* ln1g = (const float*)d_in[16]; const float* ln1b = (const float*)d_in[17];
    const float* ln2g = (const float*)d_in[18]; const float* ln2b = (const float*)d_in[19];
    const float* lnfg = (const float*)d_in[20]; const float* lnfb = (const float*)d_in[21];
    const float* headW = (const float*)d_in[22];
    float* logits = (float*)d_out;

    char* wsp = (char*)d_ws;
    auto alloc = [&](size_t bytes) -> void* {
        void* p = (void*)wsp;
        wsp += (bytes + 255) & ~(size_t)255;
        return p;
    };

    float* x  = (float*)alloc((size_t)T_SEQ * HID * 4);
    bf16* hbf = (bf16*) alloc((size_t)T_SEQ * HID * 2);
    bf16* qbf = (bf16*) alloc((size_t)T_SEQ * HID * 2);
    bf16* kbf = (bf16*) alloc((size_t)T_SEQ * KVD * 2);
    bf16* vbf = (bf16*) alloc((size_t)T_SEQ * KVD * 2);
    bf16* abf = (bf16*) alloc((size_t)T_SEQ * HID * 2);
    bf16* fbf = (bf16*) alloc((size_t)T_SEQ * FFD * 2);
    bf16* wq  = (bf16*) alloc((size_t)NLAYER * HID * HID * 2);
    bf16* wk  = (bf16*) alloc((size_t)NLAYER * HID * KVD * 2);
    bf16* wv  = (bf16*) alloc((size_t)NLAYER * HID * KVD * 2);
    bf16* wo  = (bf16*) alloc((size_t)NLAYER * HID * HID * 2);
    bf16* w1  = (bf16*) alloc((size_t)NLAYER * HID * FFD * 2);
    bf16* w2  = (bf16*) alloc((size_t)NLAYER * FFD * HID * 2);
    bf16* wh  = (bf16*) alloc((size_t)HID * VOCAB * 2);

    auto conv = [&](const float* s, bf16* d, long n) {
        convert_kernel<<<1024, 256, 0, stream>>>(s, d, n);
    };
    conv(qW,  wq, (long)NLAYER * HID * HID);
    conv(kW,  wk, (long)NLAYER * HID * KVD);
    conv(vW,  wv, (long)NLAYER * HID * KVD);
    conv(oW,  wo, (long)NLAYER * HID * HID);
    conv(m1W, w1, (long)NLAYER * HID * FFD);
    conv(m2W, w2, (long)NLAYER * FFD * HID);
    conv(headW, wh, (long)HID * VOCAB);

    embed_kernel<<<T_SEQ, 256, 0, stream>>>(input_ids, tok_emb, pos_emb, x);

    // GEMM flavors: <HAS_BIAS, GELU, HAS_RES, OUTF, OUTB>
    auto gemm_b16 = [&](const bf16* A, const bf16* B, const float* bias, bf16* oB,
                        int M, int N, int K) {   // bias, bf16 out (q/k/v projections)
        dim3 grid(N / TN, M / TM);
        gemm_bf16_kernel<true, false, false, false, true>
            <<<grid, dim3(256), 0, stream>>>(A, B, bias, nullptr, nullptr, oB, M, N, K);
    };
    auto gemm_res = [&](const bf16* A, const bf16* B, const float* bias, float* xio,
                        int M, int N, int K) {   // bias + residual, f32 out (o/m2 proj)
        dim3 grid(N / TN, M / TM);
        gemm_bf16_kernel<true, false, true, true, false>
            <<<grid, dim3(256), 0, stream>>>(A, B, bias, xio, xio, nullptr, M, N, K);
    };
    auto gemm_gelu = [&](const bf16* A, const bf16* B, const float* bias, bf16* oB,
                         int M, int N, int K) {  // bias + GELU, bf16 out (mlp up)
        dim3 grid(N / TN, M / TM);
        gemm_bf16_kernel<true, true, false, false, true>
            <<<grid, dim3(256), 0, stream>>>(A, B, bias, nullptr, nullptr, oB, M, N, K);
    };
    auto gemm_head = [&](const bf16* A, const bf16* B, float* oF,
                         int M, int N, int K) {  // no bias, f32 out (lm head)
        dim3 grid(N / TN, M / TM);
        gemm_bf16_kernel<false, false, false, true, false>
            <<<grid, dim3(256), 0, stream>>>(A, B, nullptr, nullptr, oF, nullptr, M, N, K);
    };

    for (int i = 0; i < NLAYER; ++i) {
        layernorm_kernel<<<T_SEQ, 256, 0, stream>>>(x, ln1g + i * HID, ln1b + i * HID, hbf);
        gemm_b16(hbf, wq + (size_t)i * HID * HID, qb + i * HID, qbf, T_SEQ, HID, HID);
        gemm_b16(hbf, wk + (size_t)i * HID * KVD, kb + i * KVD, kbf, T_SEQ, KVD, HID);
        gemm_b16(hbf, wv + (size_t)i * HID * KVD, vb + i * KVD, vbf, T_SEQ, KVD, HID);
        attn_kernel<<<dim3(T_SEQ / 64, NHEAD), dim3(128), 0, stream>>>(qbf, kbf, vbf,
                                                                       attn_mask, abf);
        gemm_res(abf, wo + (size_t)i * HID * HID, ob + i * HID, x, T_SEQ, HID, HID);
        layernorm_kernel<<<T_SEQ, 256, 0, stream>>>(x, ln2g + i * HID, ln2b + i * HID, hbf);
        gemm_gelu(hbf, w1 + (size_t)i * HID * FFD, m1b + i * FFD, fbf, T_SEQ, FFD, HID);
        gemm_res(fbf, w2 + (size_t)i * FFD * HID, m2b + i * HID, x, T_SEQ, HID, FFD);
    }
    layernorm_kernel<<<T_SEQ, 256, 0, stream>>>(x, lnfg, lnfb, hbf);
    gemm_head(hbf, wh, logits, T_SEQ, VOCAB, HID);
}
